// Model_50964081935589
// MI455X (gfx1250) — matrix operand order repaired and checked by reading, not compiled, not run on hardware
//
#include <hip/hip_runtime.h>
#include <hip/hip_bf16.h>

typedef _Float16 v8h  __attribute__((ext_vector_type(8)));
typedef _Float16 v16h __attribute__((ext_vector_type(16)));
typedef float    v8f  __attribute__((ext_vector_type(8)));

#define B_   32
#define T_   256
#define U_   1024
#define E_   256
#define V_   128
#define N2U  2048
#define ROWS (T_ * B_)          // 8192
#define HS   (B_ * U_)          // 32768 per-direction hidden elems

__device__ __forceinline__ v16h cat8(v8h a, v8h b) {
  return __builtin_shufflevector(a, b, 0,1,2,3,4,5,6,7,8,9,10,11,12,13,14,15);
}

// A fragment: row-major f16, p already offset to this lane's row (M = lane%16).
// Lane half hi: halves 0-7 hold K = k0+hi*8 .., halves 8-15 hold K = +16.
__device__ __forceinline__ v16h load_afrag(const _Float16* __restrict__ p,
                                           int k0, int hi) {
  int ka = k0 + hi * 8;
  v8h a0 = *(const v8h*)(p + ka);
  v8h a1 = *(const v8h*)(p + ka + 16);
  return cat8(a0, a1);
}

// B fragment: BT row-major [N][K], p offset to this lane's N row (N = lane%16).
// Lane half hi: halves 0-7 hold K = k0+hi*8 .., halves 8-15 hold K = +16.
__device__ __forceinline__ v16h load_bfrag(const _Float16* __restrict__ p,
                                           int k0, int hi) {
  int kb = k0 + hi * 8;
  v8h b0 = *(const v8h*)(p + kb);
  v8h b1 = *(const v8h*)(p + kb + 16);
  return cat8(b0, b1);
}

struct Acc4 { v8f a00, a01, a10, a11; };

// 32x32 output block = 2x2 WMMA tiles; 8 b128 loads -> 4 WMMAs per k-step.
__device__ __forceinline__ void wmma_block(const _Float16* __restrict__ Ap0,
                                           const _Float16* __restrict__ Ap1,
                                           const _Float16* __restrict__ Bp0,
                                           const _Float16* __restrict__ Bp1,
                                           int K, int hi, Acc4& acc) {
#pragma unroll 2
  for (int k0 = 0; k0 < K; k0 += 32) {
    v16h a0 = load_afrag(Ap0, k0, hi);
    v16h a1 = load_afrag(Ap1, k0, hi);
    v16h b0 = load_bfrag(Bp0, k0, hi);
    v16h b1 = load_bfrag(Bp1, k0, hi);
    acc.a00 = __builtin_amdgcn_wmma_f32_16x16x32_f16(false, a0, false, b0,
                                                     (short)0, acc.a00, false, false);
    acc.a01 = __builtin_amdgcn_wmma_f32_16x16x32_f16(false, a0, false, b1,
                                                     (short)0, acc.a01, false, false);
    acc.a10 = __builtin_amdgcn_wmma_f32_16x16x32_f16(false, a1, false, b0,
                                                     (short)0, acc.a10, false, false);
    acc.a11 = __builtin_amdgcn_wmma_f32_16x16x32_f16(false, a1, false, b1,
                                                     (short)0, acc.a11, false, false);
  }
}

__device__ __forceinline__ float sigmoidf_(float x) {
  return 1.0f / (1.0f + __expf(-x));
}

// ---------------- embedding gather: seq[t*32+b, 0:256] = emb[x[b,t]] (f16) ---
__global__ __launch_bounds__(256)
void k_embed(const int* __restrict__ x, const float* __restrict__ emb,
             _Float16* __restrict__ dst) {
  int idx = blockIdx.x * 256 + threadIdx.x;          // ROWS*E_ = 2M
  if (idx >= ROWS * E_) return;
  int r = idx >> 8, e = idx & (E_ - 1);
  int t = r >> 5, b = r & 31;
  int tok = x[b * T_ + t];
  dst[(size_t)r * N2U + e] = (_Float16)emb[(size_t)tok * E_ + e];
}

// ------- transpose+convert: dst[C x K] f16 <- src rows [r0,r0+K) of [R x C] --
__global__ __launch_bounds__(256)
void k_transpose_cvt(_Float16* __restrict__ dst, const float* __restrict__ src,
                     int C, int K, int r0) {
  int idx = blockIdx.x * 256 + threadIdx.x;
  if (idx >= C * K) return;
  int c = idx % C, k = idx / C;                       // coalesced src reads
  dst[(size_t)c * K + k] = (_Float16)src[(size_t)(r0 + k) * C + c];
}

// ---------------- h init (f32 + f16 copies) ---------------------------------
__global__ __launch_bounds__(256)
void k_init_h(float* __restrict__ H32, _Float16* __restrict__ H16,
              const float* __restrict__ h0) {
  int i = blockIdx.x * 256 + threadIdx.x;
  if (i >= HS) return;
  float v = h0[i];
  H32[i] = v;
  H16[i] = (_Float16)v;
}

// ------- generic 2x2-blocked WMMA GEMM: C[M,N] = A[M,K] @ BT[N,K]^T + bias ---
// remapT != 0: output row o reads A row (o % remapT)*B_ + o/remapT  (FC path)
__global__ __launch_bounds__(256)
void k_gemm(float* __restrict__ C, int ldc,
            const _Float16* __restrict__ A, int lda,
            const _Float16* __restrict__ BT, int ldb,
            const float* __restrict__ bias,
            int M, int N, int K, int remapT) {
  int gid  = blockIdx.x * 8 + (threadIdx.x >> 5);
  int lane = threadIdx.x & 31;
  int tilesN2 = N >> 5;
  int tm = gid / tilesN2, tn = gid - tm * tilesN2;
  if (tm * 32 >= M) return;
  int lm = lane & 15, hi = lane >> 4;
  int m0 = tm * 32, n0 = tn * 32;
  int o0 = m0 + lm, o1 = m0 + 16 + lm;
  int ar0 = remapT ? ((o0 % remapT) * B_ + o0 / remapT) : o0;
  int ar1 = remapT ? ((o1 % remapT) * B_ + o1 / remapT) : o1;
  const _Float16* Ap0 = A + (size_t)ar0 * lda;
  const _Float16* Ap1 = A + (size_t)ar1 * lda;
  const _Float16* Bp0 = BT + (size_t)(n0 + lm) * ldb;
  const _Float16* Bp1 = BT + (size_t)(n0 + 16 + lm) * ldb;
  Acc4 acc = {};
  wmma_block(Ap0, Ap1, Bp0, Bp1, K, hi, acc);
  float bv0 = bias ? bias[n0 + lm] : 0.0f;
  float bv1 = bias ? bias[n0 + 16 + lm] : 0.0f;
#pragma unroll
  for (int i = 0; i < 8; ++i) {
    int r0 = m0 + i + hi * 8;
    int r1 = r0 + 16;
    C[(size_t)r0 * ldc + n0 + lm]      = acc.a00[i] + bv0;
    C[(size_t)r0 * ldc + n0 + 16 + lm] = acc.a01[i] + bv1;
    C[(size_t)r1 * ldc + n0 + lm]      = acc.a10[i] + bv0;
    C[(size_t)r1 * ldc + n0 + 16 + lm] = acc.a11[i] + bv1;
  }
}

// ---------------- recurrent step, phase 1: gates -----------------------------
// ru_pre = GX[t] + h16 @ WghT ; sigmoid; emit rh (f16) and u (f32).
// One wave = full 32 batch rows x 32 gate cols. 128 waves: dir(2) x tn(64).
__global__ __launch_bounds__(256)
void k_step_gates(const _Float16* __restrict__ H16,
                  const _Float16* __restrict__ WGHT,
                  const float* __restrict__ GX,
                  const float* __restrict__ H32,
                  _Float16* __restrict__ RH16, float* __restrict__ U32,
                  int t_fw, int t_bw) {
  int gid  = blockIdx.x * 8 + (threadIdx.x >> 5);
  int lane = threadIdx.x & 31;
  int dir = gid >> 6;
  int tn  = gid & 63;
  int t = dir ? t_bw : t_fw;
  int lm = lane & 15, hi = lane >> 4;
  int n0 = tn * 32;
  const _Float16* Ap0 = H16 + dir * HS + (size_t)lm * U_;
  const _Float16* Ap1 = H16 + dir * HS + (size_t)(16 + lm) * U_;
  const _Float16* Bp0 = WGHT + (size_t)dir * (N2U * U_) + (size_t)(n0 + lm) * U_;
  const _Float16* Bp1 = WGHT + (size_t)dir * (N2U * U_) + (size_t)(n0 + 16 + lm) * U_;
  Acc4 acc = {};
  wmma_block(Ap0, Ap1, Bp0, Bp1, U_, hi, acc);
  const float* gx = GX + (size_t)dir * ((size_t)ROWS * N2U) + (size_t)(t * B_) * N2U;
  int na = n0 + lm, nb = n0 + 16 + lm;
#pragma unroll
  for (int i = 0; i < 8; ++i) {
    int ma = i + hi * 8;
    int mb = ma + 16;
    float v00 = sigmoidf_(acc.a00[i] + gx[(size_t)ma * N2U + na]);
    float v01 = sigmoidf_(acc.a01[i] + gx[(size_t)ma * N2U + nb]);
    float v10 = sigmoidf_(acc.a10[i] + gx[(size_t)mb * N2U + na]);
    float v11 = sigmoidf_(acc.a11[i] + gx[(size_t)mb * N2U + nb]);
    if (na < U_) {                       // whole 32-col block is reset-gate half
      int i00 = dir * HS + ma * U_ + na;
      int i01 = dir * HS + ma * U_ + nb;
      int i10 = dir * HS + mb * U_ + na;
      int i11 = dir * HS + mb * U_ + nb;
      RH16[i00] = (_Float16)(v00 * H32[i00]);
      RH16[i01] = (_Float16)(v01 * H32[i01]);
      RH16[i10] = (_Float16)(v10 * H32[i10]);
      RH16[i11] = (_Float16)(v11 * H32[i11]);
    } else {                             // update-gate half
      U32[dir * HS + ma * U_ + (na - U_)] = v00;
      U32[dir * HS + ma * U_ + (nb - U_)] = v01;
      U32[dir * HS + mb * U_ + (na - U_)] = v10;
      U32[dir * HS + mb * U_ + (nb - U_)] = v11;
    }
  }
}

// ---------------- recurrent step, phase 2: candidate + state update ----------
// c = tanh(CX[t] + rh @ WchT); h = u*h + (1-u)*c. 64 waves: dir(2) x tn(32).
__global__ __launch_bounds__(256)
void k_step_cand(const _Float16* __restrict__ RH16,
                 const _Float16* __restrict__ WCHT,
                 const float* __restrict__ CX,
                 const float* __restrict__ U32,
                 float* __restrict__ H32, _Float16* __restrict__ H16,
                 _Float16* __restrict__ Y,
                 int t_fw, int t_bw) {
  int gid  = blockIdx.x * 8 + (threadIdx.x >> 5);
  int lane = threadIdx.x & 31;
  int dir = gid >> 5;
  int tn  = gid & 31;
  int t = dir ? t_bw : t_fw;
  int lm = lane & 15, hi = lane >> 4;
  int n0 = tn * 32;
  const _Float16* Ap0 = RH16 + dir * HS + (size_t)lm * U_;
  const _Float16* Ap1 = RH16 + dir * HS + (size_t)(16 + lm) * U_;
  const _Float16* Bp0 = WCHT + (size_t)dir * (U_ * U_) + (size_t)(n0 + lm) * U_;
  const _Float16* Bp1 = WCHT + (size_t)dir * (U_ * U_) + (size_t)(n0 + 16 + lm) * U_;
  Acc4 acc = {};
  wmma_block(Ap0, Ap1, Bp0, Bp1, U_, hi, acc);
  const float* cx = CX + (size_t)dir * ((size_t)ROWS * U_) + (size_t)(t * B_) * U_;
  int na = n0 + lm, nb = n0 + 16 + lm;
#pragma unroll
  for (int i = 0; i < 8; ++i) {
    int ma = i + hi * 8;
    int mb = ma + 16;
    float pre[4] = { acc.a00[i] + cx[(size_t)ma * U_ + na],
                     acc.a01[i] + cx[(size_t)ma * U_ + nb],
                     acc.a10[i] + cx[(size_t)mb * U_ + na],
                     acc.a11[i] + cx[(size_t)mb * U_ + nb] };
    int mm[4] = { ma, ma, mb, mb };
    int nn[4] = { na, nb, na, nb };
#pragma unroll
    for (int j = 0; j < 4; ++j) {
      float c = tanhf(pre[j]);
      int ih = dir * HS + mm[j] * U_ + nn[j];
      float u = U32[ih];
      float hn = u * H32[ih] + (1.0f - u) * c;
      H32[ih] = hn;
      H16[ih] = (_Float16)hn;
      Y[(size_t)(t * B_ + mm[j]) * N2U + dir * U_ + nn[j]] = (_Float16)hn;
    }
  }
}

// ---------------- copy final states into output tuple tail -------------------
__global__ __launch_bounds__(256)
void k_save_states(const float* __restrict__ H32,
                   float* __restrict__ outFw, float* __restrict__ outBw) {
  int i = blockIdx.x * 256 + threadIdx.x;
  if (i >= HS) return;
  outFw[i] = H32[i];
  outBw[i] = H32[HS + i];
}

extern "C" void kernel_launch(void* const* d_in, const int* in_sizes, int n_in,
                              void* d_out, int out_size, void* d_ws, size_t ws_size,
                              hipStream_t stream) {
  const int*   x   = (const int*)d_in[0];
  const float* emb = (const float*)d_in[1];
  const float* fcW = (const float*)d_in[2];
  const float* fcb = (const float*)d_in[3];
  const float *Wg[3][2], *bg[3][2], *Wc[3][2], *bc[3][2], *h0[3][2];
  for (int l = 0; l < 3; ++l)
    for (int d = 0; d < 2; ++d) {
      int base = 4 + (l * 2 + d) * 5;
      Wg[l][d] = (const float*)d_in[base + 0];
      bg[l][d] = (const float*)d_in[base + 1];
      Wc[l][d] = (const float*)d_in[base + 2];
      bc[l][d] = (const float*)d_in[base + 3];
      h0[l][d] = (const float*)d_in[base + 4];
    }
  float* out = (float*)d_out;

  // ---- workspace carve (256B aligned) ----
  char* p = (char*)d_ws;
  auto carve = [&](size_t bytes) -> char* {
    char* r = p; p += (bytes + 255) & ~(size_t)255; return r;
  };
  _Float16* bufA = (_Float16*)carve((size_t)ROWS * N2U * 2);
  _Float16* bufB = (_Float16*)carve((size_t)ROWS * N2U * 2);
  float*    GX   = (float*)   carve((size_t)2 * ROWS * N2U * 4);
  float*    CX   = (float*)   carve((size_t)2 * ROWS * U_  * 4);
  _Float16* WGXT = (_Float16*)carve((size_t)2 * N2U * N2U * 2);
  _Float16* WGHT = (_Float16*)carve((size_t)2 * N2U * U_  * 2);
  _Float16* WCXT = (_Float16*)carve((size_t)2 * U_  * N2U * 2);
  _Float16* WCHT = (_Float16*)carve((size_t)2 * U_  * U_  * 2);
  _Float16* FCWT = (_Float16*)carve((size_t)V_ * N2U * 2);
  float*    H32  = (float*)   carve((size_t)2 * HS * 4);
  _Float16* H16  = (_Float16*)carve((size_t)2 * HS * 2);
  _Float16* RH16 = (_Float16*)carve((size_t)2 * HS * 2);
  float*    U32  = (float*)   carve((size_t)2 * HS * 4);
  (void)ws_size; (void)in_sizes; (void)n_in; (void)out_size;

  // ---- embedding + FC weight transpose (layer-independent) ----
  k_embed<<<(ROWS * E_ + 255) / 256, 256, 0, stream>>>(x, emb, bufA);
  k_transpose_cvt<<<(V_ * N2U + 255) / 256, 256, 0, stream>>>(FCWT, fcW, V_, N2U, 0);

  _Float16* X = bufA;
  _Float16* Y = bufB;
  const int logitsN = ROWS * V_;   // 1,048,576

  for (int l = 0; l < 3; ++l) {
    int din = (l == 0) ? E_ : N2U;
    for (int d = 0; d < 2; ++d) {
      // split + transpose + f16-convert weights: x-part and h-part
      k_transpose_cvt<<<(N2U * din + 255) / 256, 256, 0, stream>>>(
          WGXT + (size_t)d * N2U * N2U, Wg[l][d], N2U, din, 0);
      k_transpose_cvt<<<(N2U * U_ + 255) / 256, 256, 0, stream>>>(
          WGHT + (size_t)d * N2U * U_, Wg[l][d], N2U, U_, din);
      k_transpose_cvt<<<(U_ * din + 255) / 256, 256, 0, stream>>>(
          WCXT + (size_t)d * U_ * N2U, Wc[l][d], U_, din, 0);
      k_transpose_cvt<<<(U_ * U_ + 255) / 256, 256, 0, stream>>>(
          WCHT + (size_t)d * U_ * U_, Wc[l][d], U_, U_, din);
      k_init_h<<<(HS + 255) / 256, 256, 0, stream>>>(
          H32 + (size_t)d * HS, H16 + (size_t)d * HS, h0[l][d]);
      // batched input projections over all T*B rows (WMMA, 2x2 blocked)
      k_gemm<<<((ROWS / 32) * (N2U / 32)) / 8, 256, 0, stream>>>(
          GX + (size_t)d * ROWS * N2U, N2U, X, N2U,
          WGXT + (size_t)d * N2U * N2U, din, bg[l][d], ROWS, N2U, din, 0);
      k_gemm<<<((ROWS / 32) * (U_ / 32)) / 8, 256, 0, stream>>>(
          CX + (size_t)d * ROWS * U_, U_, X, N2U,
          WCXT + (size_t)d * U_ * N2U, din, bc[l][d], ROWS, U_, din, 0);
    }
    // sequential recurrence: fw time s, bw time T-1-s, both dirs per launch
    for (int s = 0; s < T_; ++s) {
      k_step_gates<<<16, 256, 0, stream>>>(H16, WGHT, GX, H32, RH16, U32,
                                           s, T_ - 1 - s);
      k_step_cand<<<8, 256, 0, stream>>>(RH16, WCHT, CX, U32, H32, H16, Y,
                                         s, T_ - 1 - s);
    }
    k_save_states<<<(HS + 255) / 256, 256, 0, stream>>>(
        H32, out + logitsN + l * HS, out + logitsN + 3 * HS + l * HS);
    _Float16* tmp = X; X = Y; Y = tmp;   // next layer consumes this output
  }

  // final FC: logits[b*T+t, :] = Y2 @ fc_W + fc_b (row remap time->batch major)
  k_gemm<<<((ROWS / 32) * (V_ / 32)) / 8, 256, 0, stream>>>(
      out, V_, X, N2U, FCWT, N2U, fcb, ROWS, V_, N2U, T_);
}
